// Elman_28295244546268
// MI455X (gfx1250) — matrix operand order repaired and checked by reading, not compiled, not run on hardware
//
#include <hip/hip_runtime.h>
#include <math.h>

#define B_ 64
#define T_ 1024
#define I_ 512
#define H_ 512
#define LDS_PITCH 520   // 512 + 8 bf16 pad -> 1040B rows, good LDS bank spread
#define NHB 4           // H-slice blocks per batch group (128 cols each)

typedef __attribute__((ext_vector_type(16))) __bf16        v16bf;
typedef __attribute__((ext_vector_type(8)))  float         v8f;
typedef __attribute__((ext_vector_type(4)))  unsigned int  v4u;
typedef __attribute__((ext_vector_type(8)))  unsigned int  v8u;

union FragAB {
    v8u   u;
    v16bf h;
    struct { v4u lo, hi; } p;
};

#if defined(__has_builtin)
#if __has_builtin(__builtin_amdgcn_global_load_async_to_lds_b128) && \
    __has_builtin(__builtin_amdgcn_s_wait_asynccnt)
#define USE_ASYNC_LDS 1
#endif
#endif

#if defined(USE_ASYNC_LDS)
// builtin signature (from hipcc diagnostic): param0 = int4-vector pointer, global AS
typedef int i4v __attribute__((vector_size(16)));
typedef __attribute__((address_space(1))) i4v* async_gp;
typedef __attribute__((address_space(3))) i4v* async_lp;
#endif

__device__ __forceinline__ unsigned short f2bf(float f) {
    unsigned int u = __float_as_uint(f);
    u += 0x7FFFu + ((u >> 16) & 1u);        // round-to-nearest-even
    return (unsigned short)(u >> 16);
}

// ---------------------------------------------------------------- init: zero h double-buffer + barrier counters
__global__ void rnn_init(unsigned int* hbuf_u32, unsigned int* cnt) {
    int i = blockIdx.x * blockDim.x + threadIdx.x;
    if (i < (2 * B_ * H_) / 2) hbuf_u32[i] = 0u;   // 2*64*512 bf16 = 32768 u32
    if (i < 4 * T_)            cnt[i]      = 0u;
}

// ---------------------------------------------------------------- pack: f32 -> bf16 weights, fused bias
__global__ void pack_weights(const float* __restrict__ Wih, const float* __restrict__ Whh,
                             const float* __restrict__ bih, const float* __restrict__ bhh,
                             unsigned short* __restrict__ WihB, unsigned short* __restrict__ WhhB,
                             float* __restrict__ bias) {
    int i = blockIdx.x * blockDim.x + threadIdx.x;
    if (i < H_ * I_) WihB[i] = f2bf(Wih[i]);
    if (i < H_ * H_) WhhB[i] = f2bf(Whh[i]);
    if (i < H_)      bias[i] = bih[i] + bhh[i];
}

// ---------------------------------------------------------------- Phase 1: Xp = X @ Wih^T + bias  -> d_out
// Block: 256 thr (8 waves). Each block: 32 rows x 512 cols; wave w owns cols [w*64, w*64+64).
__global__ __launch_bounds__(256, 1) void xp_gemm(const float* __restrict__ X,
                                                  const unsigned short* __restrict__ WihB,
                                                  const float* __restrict__ bias,
                                                  float* __restrict__ out) {
    __shared__ unsigned short aLds[32 * LDS_PITCH];
    const int tid  = threadIdx.x;
    const int wave = tid >> 5, lane = tid & 31;
    const int nl   = lane & 15, half = lane >> 4;
    const size_t row0 = (size_t)blockIdx.x * 32;

    // stage 32x512 f32 X tile as bf16 into LDS (4096 float4 / 256 thr = 16 each)
    for (int it = 0; it < 16; ++it) {
        int q  = tid + 256 * it;
        int r  = q >> 7;          // 128 float4 per row
        int c4 = q & 127;
        const float4 xv = ((const float4*)(X + (row0 + r) * I_))[c4];
        unsigned short* d = &aLds[r * LDS_PITCH + c4 * 4];
        d[0] = f2bf(xv.x); d[1] = f2bf(xv.y); d[2] = f2bf(xv.z); d[3] = f2bf(xv.w);
    }
    __syncthreads();

    const int col0 = wave * 64;
    v8f acc[2][4] = {};
    for (int s = 0; s < 16; ++s) {                 // K steps of 32
        FragAB a[2];
        #pragma unroll
        for (int mt = 0; mt < 2; ++mt) {
            const unsigned short* p = &aLds[(mt * 16 + nl) * LDS_PITCH + s * 32 + half * 16];
            a[mt].p.lo = *(const v4u*)p;           // K = half*16   .. +7
            a[mt].p.hi = *(const v4u*)(p + 8);     // K = half*16+8 .. +7
        }
        #pragma unroll
        for (int nt = 0; nt < 4; ++nt) {
            FragAB b;                              // B col n = row n of Wih, K = s*32+half*16 ..+15
            b.u = *(const v8u*)(WihB + (size_t)(col0 + nt * 16 + nl) * I_ + s * 32 + half * 16);
            #pragma unroll
            for (int mt = 0; mt < 2; ++mt)
                acc[mt][nt] = __builtin_amdgcn_wmma_f32_16x16x32_bf16(
                    false, a[mt].h, false, b.h, (short)0, acc[mt][nt], false, false);
        }
    }
    #pragma unroll
    for (int nt = 0; nt < 4; ++nt) {
        const int col = col0 + nt * 16 + nl;
        const float bs = bias[col];
        #pragma unroll
        for (int mt = 0; mt < 2; ++mt)
            #pragma unroll
            for (int v = 0; v < 8; ++v) {
                const size_t r = row0 + mt * 16 + half * 8 + v;   // D: m = half*8 + v
                out[r * H_ + col] = acc[mt][nt][v] + bs;
            }
    }
}

// ---------------------------------------------------------------- Phase 2: sequential scan
// Grid: 4 batch groups x 4 H-slices = 16 blocks, 256 thr (8 waves) each.
// Wave owns 16 H columns; its 16 Whh B-fragments (128 VGPRs) stay resident for all 1024 steps.
__global__ __launch_bounds__(256, 1) void rnn_scan(const unsigned short* __restrict__ WhhB,
                                                   unsigned short* __restrict__ hbuf,  // [2][64][512] bf16
                                                   unsigned int* __restrict__ cnt,     // [4][1024]
                                                   float* __restrict__ out) {          // holds Xp, overwritten with h
    __shared__ unsigned short hLds[16 * LDS_PITCH];
    const int tid  = threadIdx.x;
    const int wave = tid >> 5, lane = tid & 31;
    const int nl   = lane & 15, half = lane >> 4;
    const int g    = blockIdx.x / NHB;            // batch group (16 rows)
    const int j    = blockIdx.x % NHB;            // H slice (128 cols)
    const int col0 = j * 128 + wave * 16;
    const int col  = col0 + nl;

    // preload this wave's Whh B-fragments: 16 k-steps x 8 VGPRs = 128 VGPRs, resident forever
    FragAB bw[16];
    #pragma unroll
    for (int s = 0; s < 16; ++s)
        bw[s].u = *(const v8u*)(WhhB + (size_t)(col0 + nl) * H_ + s * 32 + half * 16);

    for (int t = 0; t < T_; ++t) {
        // prefetch Xp for this step early: independent of h, hide behind staging + WMMAs
        float xp[8];
        #pragma unroll
        for (int v = 0; v < 8; ++v) {
            const int b = g * 16 + half * 8 + v;
            xp[v] = out[((size_t)b * T_ + t) * H_ + col];
        }

        // stage h_{t-1} for our 16 batch rows into LDS (16KB; 1024 16B chunks / 256 thr)
        const unsigned short* hsrc = hbuf + (size_t)(t & 1) * B_ * H_ + (size_t)g * 16 * H_;
#if defined(USE_ASYNC_LDS)
        #pragma unroll
        for (int it = 0; it < 4; ++it) {
            int q = tid + 256 * it;                // 0..1023
            int r = q >> 6;                        // 64 chunks of 8 bf16 per row
            int c = q & 63;
            __builtin_amdgcn_global_load_async_to_lds_b128(
                (async_gp)(void*)(hsrc + r * H_ + c * 8),
                (async_lp)(void*)&hLds[r * LDS_PITCH + c * 8], 0, 0);
        }
        __builtin_amdgcn_s_wait_asynccnt(0);
#else
        #pragma unroll
        for (int it = 0; it < 4; ++it) {
            int q = tid + 256 * it;
            int r = q >> 6;
            int c = q & 63;
            v4u d = *(const v4u*)(hsrc + r * H_ + c * 8);
            *(v4u*)&hLds[r * LDS_PITCH + c * 8] = d;
        }
#endif
        __syncthreads();

        // recurrent GEMM: 4 independent accumulators to break the WMMA RAW chain
        v8f ac[4] = {};
        #pragma unroll
        for (int s = 0; s < 16; ++s) {
            FragAB a;
            const unsigned short* p = &hLds[nl * LDS_PITCH + s * 32 + half * 16];
            a.p.lo = *(const v4u*)p;
            a.p.hi = *(const v4u*)(p + 8);
            ac[s & 3] = __builtin_amdgcn_wmma_f32_16x16x32_bf16(
                false, a.h, false, bw[s].h, (short)0, ac[s & 3], false, false);
        }
        const v8f acc = (ac[0] + ac[1]) + (ac[2] + ac[3]);

        // epilogue: h = tanh(recurrent + Xp); write f32 out and bf16 next-h
        unsigned short* hdst = hbuf + (size_t)((t + 1) & 1) * B_ * H_;
        #pragma unroll
        for (int v = 0; v < 8; ++v) {
            const int b = g * 16 + half * 8 + v;
            const float hv = tanhf(acc[v] + xp[v]);
            out[((size_t)b * T_ + t) * H_ + col] = hv;
            hdst[(size_t)b * H_ + col] = f2bf(hv);
        }

        // release our stores, then barrier among the NHB sibling blocks of this batch group
        __builtin_amdgcn_fence(__ATOMIC_RELEASE, "agent");
        __syncthreads();
        if (tid == 0) {
            unsigned int* c = &cnt[g * T_ + t];
            __hip_atomic_fetch_add(c, 1u, __ATOMIC_RELEASE, __HIP_MEMORY_SCOPE_AGENT);
            while (__hip_atomic_load(c, __ATOMIC_ACQUIRE, __HIP_MEMORY_SCOPE_AGENT) < NHB)
                __builtin_amdgcn_s_sleep(1);
        }
        __syncthreads();
        __builtin_amdgcn_fence(__ATOMIC_ACQUIRE, "agent");   // invalidate L0 before reading siblings' h
    }
}

// ----------------------------------------------------------------
extern "C" void kernel_launch(void* const* d_in, const int* in_sizes, int n_in,
                              void* d_out, int out_size, void* d_ws, size_t ws_size,
                              hipStream_t stream) {
    const float* X   = (const float*)d_in[0];
    const float* Wih = (const float*)d_in[1];
    const float* Whh = (const float*)d_in[2];
    const float* bih = (const float*)d_in[3];
    const float* bhh = (const float*)d_in[4];
    float* out = (float*)d_out;

    char* ws = (char*)d_ws;
    unsigned short* WhhB = (unsigned short*)ws;                          // 512 KB
    unsigned short* WihB = (unsigned short*)(ws + (512u << 10));         // 512 KB
    float*          bias = (float*)(ws + (1024u << 10));                 // 2 KB
    unsigned short* hbuf = (unsigned short*)(ws + (1028u << 10));        // 128 KB
    unsigned int*   cnt  = (unsigned int*)(ws + (1028u << 10) + (128u << 10)); // 16 KB

    hipLaunchKernelGGL(rnn_init, dim3(128), dim3(256), 0, stream, (unsigned int*)hbuf, cnt);
    hipLaunchKernelGGL(pack_weights, dim3(1024), dim3(256), 0, stream,
                       Wih, Whh, bih, bhh, WihB, WhhB, bias);
    hipLaunchKernelGGL(xp_gemm, dim3((B_ * T_) / 32), dim3(256), 0, stream, X, WihB, bias, out);
    hipLaunchKernelGGL(rnn_scan, dim3(4 * NHB), dim3(256), 0, stream, WhhB, hbuf, cnt, out);
}
